// SpeakerModuleReinf_16587163697307
// MI455X (gfx1250) — matrix operand order repaired and checked by reading, not compiled, not run on hardware
//
#include <hip/hip_runtime.h>
#include <math.h>

// ---------------------------------------------------------------------------
// Problem constants (from reference): B=128, H=512, E=256, V=32000, T=32, EOS=0
// ---------------------------------------------------------------------------
#define B_    128
#define H_    512
#define E_    256
#define V_    32000
#define T_    32
#define LDA   768   // combined A row stride: [ emb(0..255) | hidden(256..767) ]
#define PITCH 520   // LDS B-panel row pitch (bf16): 4-bank skew per row -> no conflicts
#define LDSB  (64 * PITCH * 2)   // 66,560 bytes dynamic LDS for the B panel

typedef __attribute__((ext_vector_type(16))) __bf16 v16bf;
typedef __attribute__((ext_vector_type(8)))  float  v8f;

// Exact pointee type from the hipcc diagnostic for the async-LDS builtin:
// '__attribute__((__vector_size__(4 * sizeof(int)))) int'
typedef int v4i_ __attribute__((vector_size(16)));
typedef __attribute__((address_space(1))) v4i_ gv4i;   // global
typedef __attribute__((address_space(3))) v4i_ lv4i;   // LDS

union Frag { uint4 q[2]; v16bf v; };   // 32 bytes = 8 VGPRs (wave32 WMMA operand)

#if defined(__has_builtin)
#if __has_builtin(__builtin_amdgcn_global_load_async_to_lds_b128)
#define HAVE_ASYNC_LDS 1
#endif
#endif

__device__ __forceinline__ void wait_async0() {
#if defined(__has_builtin) && __has_builtin(__builtin_amdgcn_s_wait_asynccnt)
  __builtin_amdgcn_s_wait_asynccnt(0);
#else
  asm volatile("s_wait_asynccnt 0x0" ::: "memory");
#endif
}

__device__ __forceinline__ unsigned short f2bf(float f) {
  union { float f; unsigned u; } x; x.f = f;
  unsigned r = x.u + 0x7FFFu + ((x.u >> 16) & 1u);   // round-to-nearest-even
  return (unsigned short)(r >> 16);
}

// ------------------------- one-time prep kernels ---------------------------

__global__ void zero_out_k(float* __restrict__ out, long long n) {
  long long i = (long long)blockIdx.x * blockDim.x + threadIdx.x;
  long long n4 = n >> 2;
  if (i < n4) ((float4*)out)[i] = make_float4(0.f, 0.f, 0.f, 0.f);
  if (i == 0) for (long long k = n4 << 2; k < n; ++k) out[k] = 0.f;
}

__global__ void conv_wlin_k(const float* __restrict__ W_lin,
                            unsigned short* __restrict__ wbf) {
  int i = blockIdx.x * blockDim.x + threadIdx.x;      // V_*H_ threads exactly
  wbf[i] = f2bf(W_lin[i]);
}

// Pack Wrnn[n][k] = (k<256 ? W_ih[n][k] : W_hh[n][k-256]) as bf16, [H_ x LDA]
__global__ void conv_wrnn_k(const float* __restrict__ W_ih,
                            const float* __restrict__ W_hh,
                            unsigned short* __restrict__ wrnn) {
  int i = blockIdx.x * blockDim.x + threadIdx.x;      // H_*LDA threads exactly
  int n = i / LDA, k = i - n * LDA;
  float v = (k < E_) ? W_ih[n * E_ + k] : W_hh[n * H_ + (k - E_)];
  wrnn[i] = f2bf(v);
}

__global__ void init_hidden_k(const float* __restrict__ z,
                              unsigned short* __restrict__ A0) {
  int i = blockIdx.x * blockDim.x + threadIdx.x;      // B_*H_ threads exactly
  int b = i / H_, h = i - b * H_;
  A0[b * LDA + E_ + h] = f2bf(z[i]);
}

// ------------------------------ emit GEMM ----------------------------------
// logits[128, 32000] = hidden @ W_lin^T + b_lin
// One block (8 waves) owns one 64-column chunk of W_lin and ALL 8 M-tiles:
//   - stage the 64x512 bf16 W_lin panel into LDS once (async if available)
//   - wave w computes the 16x64 tile for M-tile w, B fragments from LDS
// => W_lin read from L2 exactly once per step (32 MB) instead of 8x.
__global__ __launch_bounds__(256) void emit_gemm_k(
    const unsigned short* __restrict__ Abuf,   // [128, LDA], hidden at col 256
    const unsigned short* __restrict__ wlin,   // [V_, H_] bf16
    const float* __restrict__ b_lin,
    float* __restrict__ logits)
{
  extern __shared__ unsigned char smem_raw[];
  unsigned short* Bp = (unsigned short*)smem_raw;   // [64][PITCH] bf16 panel

  const int nc  = blockIdx.x;        // 0..499 : 64-column chunk
  const int tid = threadIdx.x;

  // ---- stage B panel: 64 rows x 512 bf16 = 4096 16B chunks, 16 per thread --
  for (int c = tid; c < 4096; c += 256) {
    int row = c >> 6;                // 0..63
    int kc  = c & 63;                // 16B chunk within row
    const unsigned short* gp = wlin + ((size_t)(nc * 64 + row)) * H_ + kc * 8;
    unsigned short* lp = Bp + row * PITCH + kc * 8;
#if defined(HAVE_ASYNC_LDS)
    __builtin_amdgcn_global_load_async_to_lds_b128((gv4i*)gp, (lv4i*)lp, 0, 0);
#else
    *(uint4*)lp = *(const uint4*)gp;
#endif
  }
#if defined(HAVE_ASYNC_LDS)
  wait_async0();
#endif
  __syncthreads();

  // ---- per-wave 16x64 WMMA tile -------------------------------------------
  const int wave = tid >> 5;         // m-tile 0..7
  const int lane = tid & 31;
  const int mr = lane & 15, hi = lane >> 4;

  v8f c[4] = {v8f{0}, v8f{0}, v8f{0}, v8f{0}};
  const unsigned short* Arow = Abuf + wave * 16 * LDA + E_;   // hidden region

  for (int k0 = 0; k0 < H_; k0 += 32) {
    Frag a;
    const unsigned short* ap = Arow + mr * LDA + k0 + hi * 8;
    a.q[0] = *(const uint4*)(ap);
    a.q[1] = *(const uint4*)(ap + 16);
#pragma unroll
    for (int j = 0; j < 4; ++j) {
      Frag b;
      const unsigned short* bp = Bp + (j * 16 + mr) * PITCH + k0 + hi * 8;
      b.q[0] = *(const uint4*)(bp);
      b.q[1] = *(const uint4*)(bp + 16);
      c[j] = __builtin_amdgcn_wmma_f32_16x16x32_bf16(false, a.v, false, b.v,
                                                     (short)0, c[j], false, false);
    }
  }

#pragma unroll
  for (int j = 0; j < 4; ++j) {
    int col = nc * 64 + j * 16 + mr;
    float bl = b_lin[col];
#pragma unroll
    for (int r = 0; r < 8; ++r)
      logits[(size_t)(wave * 16 + hi * 8 + r) * V_ + col] = c[j][r] + bl;
  }
}

// ----------------------- emit reduce / decode step -------------------------
// Per row b: argmax (first occurrence), lp = -log(sum exp(l - max)), update
// mask / log_p_sent recurrence, scatter one-hot into d_out[t].
__global__ __launch_bounds__(256) void reduce_emit_k(
    const float* __restrict__ logits, int t,
    int* __restrict__ sample, float* __restrict__ mask, float* __restrict__ lps,
    float* __restrict__ out)
{
  __shared__ float sv[256];
  __shared__ int   si[256];
  const int b = blockIdx.x, tid = threadIdx.x;
  const float* row = logits + (size_t)b * V_;

  float bm = -3.4e38f; int bi = 0;
  for (int i = tid; i < V_; i += 256) {
    float v = row[i];
    if (v > bm) { bm = v; bi = i; }
  }
  sv[tid] = bm; si[tid] = bi; __syncthreads();
  for (int s = 128; s > 0; s >>= 1) {
    if (tid < s) {
      float ov = sv[tid + s]; int oi = si[tid + s];
      if (ov > sv[tid] || (ov == sv[tid] && oi < si[tid])) { sv[tid] = ov; si[tid] = oi; }
    }
    __syncthreads();
  }
  const float M = sv[0]; const int am = si[0];
  __syncthreads();

  float se = 0.f;
  for (int i = tid; i < V_; i += 256) se += expf(row[i] - M);
  sv[tid] = se; __syncthreads();
  for (int s = 128; s > 0; s >>= 1) {
    if (tid < s) sv[tid] += sv[tid + s];
    __syncthreads();
  }

  if (tid == 0) {
    float lp = -logf(sv[0]);                 // log_p at argmax token
    float l, m;
    if (t == 0) { l = lp;              m = (am == 0) ? 0.f : 1.f; }
    else {
      float mo = mask[b];
      l = lps[b] + mo * lp;
      m = mo * ((am == 0) ? 0.f : 1.f);
    }
    lps[b] = l; mask[b] = m; sample[b] = am;
    out[(size_t)t * B_ * V_ + (size_t)b * V_ + am] = 1.0f;
    if (t == T_ - 1) out[(size_t)T_ * B_ * V_ + b] = l;
  }
}

// ------------------------- one-hot embedding gather ------------------------
// A[b][e] = bf16(W_emb[e][sample_b] + b_emb[e]); grid 128 x 256
__global__ void gather_emb_k(const float* __restrict__ W_emb,
                             const float* __restrict__ b_emb,
                             const int* __restrict__ sample,
                             unsigned short* __restrict__ Abuf)
{
  int b = blockIdx.x, e = threadIdx.x;
  int s = sample[b];
  Abuf[b * LDA + e] = f2bf(W_emb[(size_t)e * V_ + s] + b_emb[e]);
}

// ------------------------------ RNN cell GEMM ------------------------------
// hid_next = tanh( [emb|hid] @ [W_ih|W_hh]^T + b_ih + b_hh ), written as bf16
// into Anext hidden region. M=128, N=512, K=768 -> 64 waves (8 blocks x 256).
__global__ __launch_bounds__(256) void cell_gemm_k(
    const unsigned short* __restrict__ Acur,   // [128, LDA]
    const unsigned short* __restrict__ wrnn,   // [H_, LDA] bf16
    const float* __restrict__ b_ih, const float* __restrict__ b_hh,
    unsigned short* __restrict__ Anext)
{
  int wave = blockIdx.x * (blockDim.x >> 5) + (threadIdx.x >> 5);
  int lane = threadIdx.x & 31;
  int mt = wave & 7;       // 8 M tiles
  int nc = wave >> 3;      // 8 chunks of 64 -> N=512
  const int mr = lane & 15, hi = lane >> 4;

  v8f c[4] = {v8f{0}, v8f{0}, v8f{0}, v8f{0}};
  const unsigned short* Arow = Acur + mt * 16 * LDA;
  const unsigned short* Bm   = wrnn + nc * 64 * LDA;
  for (int k0 = 0; k0 < LDA; k0 += 32) {
    Frag a;
    const unsigned short* ap = Arow + mr * LDA + k0 + hi * 8;
    a.q[0] = *(const uint4*)(ap);
    a.q[1] = *(const uint4*)(ap + 16);
#pragma unroll
    for (int j = 0; j < 4; ++j) {
      Frag b;
      const unsigned short* bp = Bm + (j * 16 + mr) * LDA + k0 + hi * 8;
      b.q[0] = *(const uint4*)(bp);
      b.q[1] = *(const uint4*)(bp + 16);
      c[j] = __builtin_amdgcn_wmma_f32_16x16x32_bf16(false, a.v, false, b.v,
                                                     (short)0, c[j], false, false);
    }
  }
#pragma unroll
  for (int j = 0; j < 4; ++j) {
    int n = nc * 64 + j * 16 + mr;
    float bias = b_ih[n] + b_hh[n];
#pragma unroll
    for (int r = 0; r < 8; ++r) {
      int row = mt * 16 + hi * 8 + r;
      Anext[row * LDA + E_ + n] = f2bf(tanhf(c[j][r] + bias));
    }
  }
}

// ---------------------------------------------------------------------------
extern "C" void kernel_launch(void* const* d_in, const int* in_sizes, int n_in,
                              void* d_out, int out_size, void* d_ws, size_t ws_size,
                              hipStream_t stream) {
  const float* z     = (const float*)d_in[0];
  const float* W_emb = (const float*)d_in[1];
  const float* b_emb = (const float*)d_in[2];
  const float* W_ih  = (const float*)d_in[3];
  const float* b_ih  = (const float*)d_in[4];
  const float* W_hh  = (const float*)d_in[5];
  const float* b_hh  = (const float*)d_in[6];
  const float* W_lin = (const float*)d_in[7];
  const float* b_lin = (const float*)d_in[8];
  (void)in_sizes; (void)n_in; (void)ws_size;

  // workspace layout (256B aligned slabs), ~50.7 MB total
  char* ws = (char*)d_ws;
  size_t off = 0;
  auto alloc = [&](size_t bytes) -> void* {
    void* p = ws + off;
    off = (off + bytes + 255) & ~(size_t)255;
    return p;
  };
  unsigned short* wlin_bf = (unsigned short*)alloc((size_t)V_ * H_ * 2);
  unsigned short* wrnn_bf = (unsigned short*)alloc((size_t)H_ * LDA * 2);
  float*          logits  = (float*)alloc((size_t)B_ * V_ * 4);
  unsigned short* A0      = (unsigned short*)alloc((size_t)B_ * LDA * 2);
  unsigned short* A1      = (unsigned short*)alloc((size_t)B_ * LDA * 2);
  int*            sample  = (int*)alloc(B_ * 4);
  float*          maskp   = (float*)alloc(B_ * 4);
  float*          lpsp    = (float*)alloc(B_ * 4);

  // 1) zero the (poisoned) output: T*B*V one-hot plane + B log-probs
  {
    long long n = (long long)out_size;
    long long n4 = n >> 2;
    int blocks = (int)((n4 + 255) / 256);
    zero_out_k<<<blocks, 256, 0, stream>>>((float*)d_out, n);
  }
  // 2) bf16 weight prep (L2-resident afterwards: 33 MB + 0.8 MB << 192 MB L2)
  conv_wlin_k<<<(V_ * H_) / 256, 256, 0, stream>>>(W_lin, wlin_bf);
  conv_wrnn_k<<<(H_ * LDA) / 256, 256, 0, stream>>>(W_ih, W_hh, wrnn_bf);
  init_hidden_k<<<(B_ * H_) / 256, 256, 0, stream>>>(z, A0);

  // 3) sequential greedy decode, T steps
  unsigned short* Acur = A0;
  unsigned short* Anxt = A1;
  for (int t = 0; t < T_; ++t) {
    emit_gemm_k<<<500, 256, LDSB, stream>>>(Acur, wlin_bf, b_lin, logits);
    reduce_emit_k<<<B_, 256, 0, stream>>>(logits, t, sample, maskp, lpsp,
                                          (float*)d_out);
    if (t < T_ - 1) {
      gather_emb_k<<<B_, E_, 0, stream>>>(W_emb, b_emb, sample, Acur);
      cell_gemm_k<<<8, 256, 0, stream>>>(Acur, wrnn_bf, b_ih, b_hh, Anxt);
      unsigned short* tmp = Acur; Acur = Anxt; Anxt = tmp;
    }
  }
}
